// Offset_Learning_35871566856393
// MI455X (gfx1250) — matrix operand order repaired and checked by reading, not compiled, not run on hardware
//
#include <hip/hip_runtime.h>
#include <hip/hip_fp16.h>

typedef __attribute__((ext_vector_type(16))) _Float16 v16h;
typedef __attribute__((ext_vector_type(8)))  _Float16 v8h;
typedef __attribute__((ext_vector_type(8)))  float    v8f;

#define BDIM 256
static constexpr int B_  = 8;
static constexpr int C_  = 512;
static constexpr int N_  = 128 * 128;   // H*W = 16384
static constexpr int K_  = 150;
static constexpr int KP_ = 160;         // K padded to multiple of 32

// ---------------------------------------------------------------- casts
__global__ void cast_f32_to_f16(const float* __restrict__ src,
                                _Float16* __restrict__ dst, int n) {
  for (int i = blockIdx.x * blockDim.x + threadIdx.x; i < n;
       i += gridDim.x * blockDim.x)
    dst[i] = (_Float16)src[i];
}

// x[b][c][n] (f32) -> imgT[b][n][c] (f16), 32x32 LDS tiles
__global__ void transpose_cast_kernel(const float* __restrict__ x,
                                      _Float16* __restrict__ imgT) {
  __shared__ float t[32][33];
  int b  = blockIdx.z;
  int c0 = blockIdx.x * 32, n0 = blockIdx.y * 32;
  int tx = threadIdx.x, ty = threadIdx.y;       // 32 x 8
  const float* xp = x + (size_t)b * C_ * N_;
#pragma unroll
  for (int j = 0; j < 4; j++) {
    int c = c0 + ty + j * 8;
    t[ty + j * 8][tx] = xp[(size_t)c * N_ + n0 + tx];
  }
  __syncthreads();
  _Float16* op = imgT + (size_t)b * N_ * C_;
#pragma unroll
  for (int j = 0; j < 4; j++) {
    int n = n0 + ty + j * 8;
    op[(size_t)n * C_ + c0 + tx] = (_Float16)t[tx][ty + j * 8];
  }
}

// cls_repr -> clsb [KP_][C_] and clsT [C_][KP_], zero padded rows/cols
__global__ void cls_cast_kernel(const float* __restrict__ cls,
                                _Float16* __restrict__ clsb,
                                _Float16* __restrict__ clsT) {
  int idx = blockIdx.x * blockDim.x + threadIdx.x;
  if (idx >= KP_ * C_) return;
  int k = idx / C_, c = idx % C_;
  float v = (k < K_) ? cls[k * C_ + c] : 0.f;
  clsb[k * C_ + c]  = (_Float16)v;
  clsT[c * KP_ + k] = (_Float16)v;
}

// ---------------------------------------------------------------- softmaxes
// softmax over n (row) of coupled[b][k][n] -> f16 attn[b][k][n], pad rows = 0
__global__ void row_softmax_kernel(const float* __restrict__ coupled,
                                   _Float16* __restrict__ attn) {
  int k = blockIdx.x, b = blockIdx.y, tid = threadIdx.x;
  _Float16* out = attn + ((size_t)b * KP_ + k) * N_;
  if (k >= K_) {
    for (int i = tid; i < N_; i += BDIM) out[i] = (_Float16)0.f;
    return;
  }
  const float* row = coupled + ((size_t)b * K_ + k) * N_;
  __shared__ float red[BDIM];
  float mx = -1e30f;
  for (int i = tid; i < N_; i += BDIM) mx = fmaxf(mx, row[i]);
  red[tid] = mx; __syncthreads();
  for (int s = BDIM / 2; s > 0; s >>= 1) {
    if (tid < s) red[tid] = fmaxf(red[tid], red[tid + s]);
    __syncthreads();
  }
  mx = red[0]; __syncthreads();
  float sum = 0.f;
  for (int i = tid; i < N_; i += BDIM) sum += __expf(row[i] - mx);
  red[tid] = sum; __syncthreads();
  for (int s = BDIM / 2; s > 0; s >>= 1) {
    if (tid < s) red[tid] += red[tid + s];
    __syncthreads();
  }
  float rs = 1.f / red[0];
  for (int i = tid; i < N_; i += BDIM)
    out[i] = (_Float16)(__expf(row[i] - mx) * rs);
}

// softmax over k (col) of coupled[b][k][n] -> f16 attnT[b][n][k], pad = 0
__global__ void col_softmax_kernel(const float* __restrict__ coupled,
                                   _Float16* __restrict__ attnT) {
  int n = blockIdx.x * BDIM + threadIdx.x;
  int b = blockIdx.y;
  const float* base = coupled + (size_t)b * K_ * N_ + n;
  float mx = -1e30f;
  for (int k = 0; k < K_; k++) mx = fmaxf(mx, base[(size_t)k * N_]);
  float sum = 0.f;
  for (int k = 0; k < K_; k++) sum += __expf(base[(size_t)k * N_] - mx);
  float rs = 1.f / sum;
  _Float16* o = attnT + ((size_t)b * N_ + n) * KP_;
  for (int k = 0; k < K_; k++)
    o[k] = (_Float16)(__expf(base[(size_t)k * N_] - mx) * rs);
  for (int k = K_; k < KP_; k++) o[k] = (_Float16)0.f;
}

// ---------------------------------------------------------------- WMMA GEMM
// OUT[M,N] = A[M,Kd] * BT[N,Kd]^T ; A,BT f16 row-major; f32 accumulate.
// Block tile 128x64: 8 waves in 4x2 grid, each wave owns 32x32
// (2 A-fragments x 2 B-fragments -> 4 WMMA per k-step, 8 b128 loads).
// EPI: 0 = store f32, 1 = store f16, 2 = store f16 of (acc + ADD[row][col])
template <int EPI>
__global__ __launch_bounds__(256) void gemm_wmma_f16(
    const _Float16* __restrict__ A, size_t sA, int lda, int Malloc, int Mv,
    const _Float16* __restrict__ BT, size_t sBT, int ldbt, int Nalloc, int Nv,
    int Kd, void* __restrict__ OUT, size_t sO, int ldo,
    const _Float16* __restrict__ ADD, size_t sAdd, int ldadd) {
  const int b = blockIdx.z;
  const _Float16* Ab  = A + (size_t)b * sA;
  const _Float16* BTb = BT + (size_t)b * sBT;
  int tid = threadIdx.x, wid = tid >> 5, lane = tid & 31;
  int hl = lane >> 4, l16 = lane & 15;       // lane half, lane%16
  int wm = wid & 3, wn = wid >> 2;           // 4x2 wave grid -> 128x64 block
  int m0 = blockIdx.y * 128 + wm * 32;
  int n0 = blockIdx.x * 64 + wn * 32;

  int ar0 = m0 + l16;      if (ar0 > Malloc - 1) ar0 = Malloc - 1;
  int ar1 = m0 + 16 + l16; if (ar1 > Malloc - 1) ar1 = Malloc - 1;
  const _Float16* Ap0 = Ab + (size_t)ar0 * lda;
  const _Float16* Ap1 = Ab + (size_t)ar1 * lda;
  int bc0 = n0 + l16;      if (bc0 > Nalloc - 1) bc0 = Nalloc - 1;
  int bc1 = n0 + 16 + l16; if (bc1 > Nalloc - 1) bc1 = Nalloc - 1;
  const _Float16* Bp0 = BTb + (size_t)bc0 * ldbt;
  const _Float16* Bp1 = BTb + (size_t)bc1 * ldbt;

  v8f acc00 = {}; v8f acc01 = {}; v8f acc10 = {}; v8f acc11 = {};
  for (int k = 0; k < Kd; k += 32) {
    if (k + 128 < Kd) {                      // global_prefetch_b8 hints
      __builtin_prefetch(Ap0 + k + 128, 0, 1);
      __builtin_prefetch(Ap1 + k + 128, 0, 1);
      __builtin_prefetch(Bp0 + k + 128, 0, 1);
      __builtin_prefetch(Bp1 + k + 128, 0, 1);
    }
    // A 16x32 f16 fragment: lane holds K = hl*8+e and 16+hl*8+e (ISA layout)
    v8h a0lo = *(const v8h*)(Ap0 + k + hl * 8);
    v8h a0hi = *(const v8h*)(Ap0 + k + 16 + hl * 8);
    v8h a1lo = *(const v8h*)(Ap1 + k + hl * 8);
    v8h a1hi = *(const v8h*)(Ap1 + k + 16 + hl * 8);
    // B 32x16 f16 fragment: lane holds K = hl*8+e and 16+hl*8+e (same as A)
    v8h b0l = *(const v8h*)(Bp0 + k + hl * 8);
    v8h b0h = *(const v8h*)(Bp0 + k + 16 + hl * 8);
    v8h b1l = *(const v8h*)(Bp1 + k + hl * 8);
    v8h b1h = *(const v8h*)(Bp1 + k + 16 + hl * 8);
    v16h av0, av1, bv0, bv1;
#pragma unroll
    for (int i = 0; i < 8; i++) {
      av0[i] = a0lo[i]; av0[i + 8] = a0hi[i];
      av1[i] = a1lo[i]; av1[i + 8] = a1hi[i];
      bv0[i] = b0l[i];  bv0[i + 8] = b0h[i];
      bv1[i] = b1l[i];  bv1[i + 8] = b1h[i];
    }
    acc00 = __builtin_amdgcn_wmma_f32_16x16x32_f16(false, av0, false, bv0,
                                                   (short)0, acc00, false, false);
    acc01 = __builtin_amdgcn_wmma_f32_16x16x32_f16(false, av0, false, bv1,
                                                   (short)0, acc01, false, false);
    acc10 = __builtin_amdgcn_wmma_f32_16x16x32_f16(false, av1, false, bv0,
                                                   (short)0, acc10, false, false);
    acc11 = __builtin_amdgcn_wmma_f32_16x16x32_f16(false, av1, false, bv1,
                                                   (short)0, acc11, false, false);
  }
  int c0 = n0 + l16, c1 = n0 + 16 + l16;
#pragma unroll
  for (int r = 0; r < 8; r++) {
    int row0 = m0 + hl * 8 + r;              // C/D layout: M = r + half*8
    int row1 = row0 + 16;
#pragma unroll
    for (int t = 0; t < 2; t++) {
      int row = t ? row1 : row0;
      float v0 = t ? acc10[r] : acc00[r];
      float v1 = t ? acc11[r] : acc01[r];
      if (row >= Mv) continue;
      if (EPI == 0) {
        float* O = (float*)OUT + (size_t)b * sO + (size_t)row * ldo;
        if (c0 < Nv) O[c0] = v0;
        if (c1 < Nv) O[c1] = v1;
      } else if (EPI == 1) {
        _Float16* O = (_Float16*)OUT + (size_t)b * sO + (size_t)row * ldo;
        if (c0 < Nv) O[c0] = (_Float16)v0;
        if (c1 < Nv) O[c1] = (_Float16)v1;
      } else {
        _Float16* O = (_Float16*)OUT + (size_t)b * sO + (size_t)row * ldo;
        const _Float16* Ad = ADD + (size_t)b * sAdd + (size_t)row * ldadd;
        if (c0 < Nv) O[c0] = (_Float16)(v0 + (float)Ad[c0]);
        if (c1 < Nv) O[c1] = (_Float16)(v1 + (float)Ad[c1]);
      }
    }
  }
}

// ------------------------------------------------- LayerNorm(K) + transpose
__global__ __launch_bounds__(256) void layernorm_out_kernel(
    const float* __restrict__ masks, const float* __restrict__ gamma,
    const float* __restrict__ beta, float* __restrict__ out) {
  __shared__ float tile[64 * K_];
  __shared__ float muS[64], rsS[64];
  int b = blockIdx.y, n0 = blockIdx.x * 64, tid = threadIdx.x;
  const float* base = masks + ((size_t)b * N_ + n0) * K_;
  for (int i = tid; i < 64 * K_; i += BDIM) tile[i] = base[i];
  __syncthreads();
  if (tid < 64) {
    const float* rp = &tile[tid * K_];
    float s = 0.f, ss = 0.f;
    for (int k = 0; k < K_; k++) { float v = rp[k]; s += v; ss += v * v; }
    float mu = s * (1.f / K_);
    float var = ss * (1.f / K_) - mu * mu;   // population variance (jnp.var)
    muS[tid] = mu;
    rsS[tid] = rsqrtf(var + 1e-5f);
  }
  __syncthreads();
  for (int idx = tid; idx < K_ * 64; idx += BDIM) {
    int k = idx >> 6, r = idx & 63;
    float v = (tile[r * K_ + k] - muS[r]) * rsS[r] * gamma[k] + beta[k];
    out[((size_t)b * K_ + k) * N_ + n0 + r] = v;   // coalesced over r
  }
}

// ---------------------------------------------------------------- launcher
extern "C" void kernel_launch(void* const* d_in, const int* in_sizes, int n_in,
                              void* d_out, int out_size, void* d_ws,
                              size_t ws_size, hipStream_t stream) {
  const float* x        = (const float*)d_in[0];
  const float* cls_repr = (const float*)d_in[1];
  const float* W_cls    = (const float*)d_in[2];
  const float* W_feat   = (const float*)d_in[3];
  const float* ln_gamma = (const float*)d_in[4];
  const float* ln_beta  = (const float*)d_in[5];
  float* out = (float*)d_out;

  char* ws = (char*)d_ws;
  size_t off = 0;
  auto alloc = [&](size_t bytes) {
    void* p = ws + off;
    off += (bytes + 255) & ~(size_t)255;
    return p;
  };
  _Float16* imgb    = (_Float16*)alloc((size_t)B_ * N_ * C_ * 2); // img [b][n][c]
  _Float16* xb      = (_Float16*)alloc((size_t)B_ * C_ * N_ * 2); // img^T; reused as ub
  float*    coupled = (float*)   alloc((size_t)B_ * K_ * N_ * 4); // reused as masks
  _Float16* clsattn = (_Float16*)alloc((size_t)B_ * KP_ * N_ * 2);
  _Float16* posT    = (_Float16*)alloc((size_t)B_ * N_ * KP_ * 2);
  _Float16* afb     = (_Float16*)alloc((size_t)B_ * N_ * C_ * 2);
  _Float16* t_kc    = (_Float16*)alloc((size_t)B_ * KP_ * C_ * 2);
  _Float16* acb     = (_Float16*)alloc((size_t)B_ * KP_ * C_ * 2);
  _Float16* clsb    = (_Float16*)alloc((size_t)KP_ * C_ * 2);
  _Float16* clsT    = (_Float16*)alloc((size_t)C_ * KP_ * 2);
  _Float16* Wcb     = (_Float16*)alloc((size_t)C_ * C_ * 2);
  _Float16* Wfb     = (_Float16*)alloc((size_t)C_ * C_ * 2);
  _Float16* ub    = xb;       // xb dead after GEMM2
  float*    masks = coupled;  // coupled dead after softmaxes

  // ---- precision casts / transposes
  cast_f32_to_f16<<<4096, BDIM, 0, stream>>>(x, xb, B_ * C_ * N_);
  cast_f32_to_f16<<<1024, BDIM, 0, stream>>>(W_cls, Wcb, C_ * C_);
  cast_f32_to_f16<<<1024, BDIM, 0, stream>>>(W_feat, Wfb, C_ * C_);
  transpose_cast_kernel<<<dim3(C_ / 32, N_ / 32, B_), dim3(32, 8), 0, stream>>>(
      x, imgb);
  cls_cast_kernel<<<(KP_ * C_ + BDIM - 1) / BDIM, BDIM, 0, stream>>>(
      cls_repr, clsb, clsT);

  // ---- GEMM1: coupled[k][n] = cls * img^T   (A=clsb, BT=imgb)
  gemm_wmma_f16<0><<<dim3(N_ / 64, 2, B_), BDIM, 0, stream>>>(
      clsb, 0, C_, KP_, K_,
      imgb, (size_t)N_ * C_, C_, N_, N_,
      C_, coupled, (size_t)K_ * N_, N_, nullptr, 0, 0);

  // ---- softmaxes (f32 in, f16 out)
  row_softmax_kernel<<<dim3(KP_, B_), BDIM, 0, stream>>>(coupled, clsattn);
  col_softmax_kernel<<<dim3(N_ / BDIM, B_), BDIM, 0, stream>>>(coupled, posT);

  // ---- GEMM2: t[k][c] = cls_attn * img     (A=clsattn, BT=xb [c][n])
  gemm_wmma_f16<1><<<dim3(C_ / 64, 2, B_), BDIM, 0, stream>>>(
      clsattn, (size_t)KP_ * N_, N_, KP_, KP_,
      xb, (size_t)C_ * N_, N_, C_, C_,
      N_, t_kc, (size_t)KP_ * C_, C_, nullptr, 0, 0);

  // ---- GEMM2b: aligned_cls = t * W_cls^T + cls
  gemm_wmma_f16<2><<<dim3(C_ / 64, 2, B_), BDIM, 0, stream>>>(
      t_kc, (size_t)KP_ * C_, C_, KP_, KP_,
      Wcb, 0, C_, C_, C_,
      C_, acb, (size_t)KP_ * C_, C_, clsb, 0, C_);

  // ---- GEMM3: u[n][c] = pos_attn^T * cls   (A=posT, BT=clsT [c][k])
  gemm_wmma_f16<1><<<dim3(C_ / 64, N_ / 128, B_), BDIM, 0, stream>>>(
      posT, (size_t)N_ * KP_, KP_, N_, N_,
      clsT, 0, KP_, C_, C_,
      KP_, ub, (size_t)N_ * C_, C_, nullptr, 0, 0);

  // ---- GEMM4: aligned_feat = u * W_feat^T + img
  gemm_wmma_f16<2><<<dim3(C_ / 64, N_ / 128, B_), BDIM, 0, stream>>>(
      ub, (size_t)N_ * C_, C_, N_, N_,
      Wfb, 0, C_, C_, C_,
      C_, afb, (size_t)N_ * C_, C_, imgb, (size_t)N_ * C_, C_);

  // ---- GEMM5: masks[n][k] = aligned_feat * aligned_cls^T
  gemm_wmma_f16<0><<<dim3(3, N_ / 128, B_), BDIM, 0, stream>>>(
      afb, (size_t)N_ * C_, C_, N_, N_,
      acb, (size_t)KP_ * C_, C_, KP_, K_,
      C_, masks, (size_t)N_ * K_, K_, nullptr, 0, 0);

  // ---- LayerNorm over K + transpose to [b][K][H][W]
  layernorm_out_kernel<<<dim3(N_ / 64, B_), BDIM, 0, stream>>>(
      masks, ln_gamma, ln_beta, out);
}
